// GResNet_19799799234726
// MI455X (gfx1250) — matrix operand order repaired and checked by reading, not compiled, not run on hardware
//
#include <hip/hip_runtime.h>
#include <hip/hip_bf16.h>
#include <stdint.h>

// ---------------------------------------------------------------------------
// GCN-ResNet forward for MI455X (gfx1250, wave32).
//  - GEMMs via v_wmma_f32_16x16x32_bf16; weights pre-swizzled into B-fragment
//    order so B loads are 2x global_load_b128 per tile (L2-resident), A staged
//    through LDS branch-free (clamped indices, packed bf16, ds_store_b64/b32)
//  - edge scatter via global fp32 atomics: per-graph agg (20.5MB) fits in L2
// Workspace requirement ~65 MB.
// ---------------------------------------------------------------------------

typedef __attribute__((ext_vector_type(16))) __bf16 v16bf;
typedef __attribute__((ext_vector_type(8)))  float  v8f;

#define DEVINL static __device__ __forceinline__

DEVINL unsigned short f2bf(float f) {           // fp32 -> bf16 RNE
    unsigned int u = __float_as_uint(f);
    unsigned int r = u + 0x7FFFu + ((u >> 16) & 1u);
    return (unsigned short)(r >> 16);
}
DEVINL unsigned int pack_bf2(float lo, float hi) {
    return (unsigned int)f2bf(lo) | ((unsigned int)f2bf(hi) << 16);
}

union BFrag { uint4 q[2]; unsigned int u[8]; v16bf v; };

// ---------------------------------------------------------------------------
// Weight swizzle: W[K,N] fp32 -> B-fragment-ordered packed bf16 dwords.
// Layout: dword[((kb*(N/16) + nt)*32 + lane)*8 + j] holds
//   K = kb*32 + (lane/16)*16 + 2j (+1 in high half),  n = nt*16 + lane%16.
// K padded to multiple of 32 with zeros.
// ---------------------------------------------------------------------------
__global__ void swz_w_k(const float* __restrict__ W, unsigned int* __restrict__ out,
                        int K, int N, int total) {
    int t = blockIdx.x * blockDim.x + threadIdx.x;
    if (t >= total) return;
    int j    = t & 7;
    int lane = (t >> 3) & 31;
    int rest = t >> 8;
    int n16  = N >> 4;
    int nt   = rest % n16;
    int kb   = rest / n16;
    int half = lane >> 4, l16 = lane & 15;
    int k = kb * 32 + half * 16 + 2 * j;
    int n = nt * 16 + l16;
    float v0 = (k     < K) ? W[(size_t)k       * N + n] : 0.f;
    float v1 = (k + 1 < K) ? W[(size_t)(k + 1) * N + n] : 0.f;
    out[t] = pack_bf2(v0, v1);
}

// ---------------------------------------------------------------------------
// WMMA GEMM:  C[M,N] = A[M,K](fp32) * Wsw(pre-swizzled bf16) (+ bias[N])
// Block = 256 threads (8 waves). WG tile: 128(M) x 64(N) x 32(K).
// Wave w owns rows [w*16, w*16+16) and 4 N-subtiles -> 4 WMMAs / k-step.
// ALIGNED4: K % 4 == 0 and A rows 16B-aligned (true for K=256 layers).
// ---------------------------------------------------------------------------
template <bool ALIGNED4>
__global__ __launch_bounds__(256)
void gemm_v2(const float* __restrict__ A,
             const unsigned int* __restrict__ Wsw,
             const float* __restrict__ bias,
             float* __restrict__ C,
             int M, int K, int N)
{
    __shared__ unsigned short As[128][36];   // [m][k], stride 36 -> rows 8B aligned

    const int tid  = threadIdx.x;
    const int lane = tid & 31;
    const int w    = tid >> 5;          // wave 0..7 -> M strip
    const int m0   = blockIdx.x * 128;
    const int nt0  = blockIdx.y * 4;    // first N-subtile (16 wide each)
    const int half = lane >> 4;
    const int l16  = lane & 15;
    const int n16  = N >> 4;

    v8f acc[4];
    #pragma unroll
    for (int t = 0; t < 4; ++t) acc[t] = v8f{0.f,0.f,0.f,0.f,0.f,0.f,0.f,0.f};

    const int nKb = (K + 31) >> 5;
    for (int kb = 0; kb < nKb; ++kb) {
        const int k0 = kb * 32;

        // ---- stage A tile (128 x 32) fp32 -> packed bf16, branch-free ----
        if (ALIGNED4) {
            // 1024 quads, 4 per thread: float4 load + ds_store_b64
            #pragma unroll
            for (int i = 0; i < 4; ++i) {
                int p = tid + i * 256;          // quad index
                int r = p >> 3;                 // row 0..127 (8 quads/row)
                int c = (p & 7) * 4;            // k offset 0,4,..28
                int gm = m0 + r; gm = gm < M ? gm : (M - 1);   // clamp: rows >= M unused
                const float4 v = *(const float4*)(A + (size_t)gm * K + (k0 + c));
                uint2 pk = { pack_bf2(v.x, v.y), pack_bf2(v.z, v.w) };
                *(uint2*)&As[r][c] = pk;
            }
        } else {
            // generic: 2048 pairs, 8 per thread, clamped index + select for K tail
            #pragma unroll
            for (int i = 0; i < 8; ++i) {
                int p = tid + i * 256;          // pair index
                int r = p >> 4;                 // row 0..127 (16 pairs/row)
                int c = (p & 15) * 2;           // k offset 0,2,..30
                int gm = m0 + r; gm = gm < M ? gm : (M - 1);
                int gk = k0 + c;
                const float* ap = A + (size_t)gm * K;
                int i0 = gk     < K ? gk     : (K - 1);
                int i1 = gk + 1 < K ? gk + 1 : (K - 1);
                float v0 = ap[i0]; v0 = (gk     < K) ? v0 : 0.f;
                float v1 = ap[i1]; v1 = (gk + 1 < K) ? v1 : 0.f;
                *(unsigned int*)&As[r][c] = pack_bf2(v0, v1);
            }
        }
        __syncthreads();

        // ---- A fragment (16x32 bf16 layout) via 32-bit ds loads ----
        union { unsigned int u[8]; v16bf v; } af;
        const unsigned short* arow = &As[w * 16 + l16][0];
        #pragma unroll
        for (int j = 0; j < 4; ++j)
            af.u[j]     = *(const unsigned int*)(arow + (half * 16 + 2 * j));
        #pragma unroll
        for (int j = 0; j < 4; ++j)
            af.u[4 + j] = *(const unsigned int*)(arow + (half * 16 + 8 + 2 * j));

        // ---- 4 B fragments direct from pre-swizzled global (L2) + WMMA ----
        #pragma unroll
        for (int t = 0; t < 4; ++t) {
            BFrag bf;
            const uint4* bp = (const uint4*)
                (Wsw + (((size_t)kb * n16 + (nt0 + t)) * 32 + lane) * 8);
            bf.q[0] = bp[0];
            bf.q[1] = bp[1];
            acc[t] = __builtin_amdgcn_wmma_f32_16x16x32_bf16(
                         false, af.v, false, bf.v, (short)0, acc[t], false, false);
        }
        __syncthreads();
    }

    // ---- epilogue: C/D layout -> VGPR r: M = r + half*8; N = lane%16 ----
    #pragma unroll
    for (int t = 0; t < 4; ++t) {
        int n = (nt0 + t) * 16 + l16;
        float bv = bias ? bias[n] : 0.f;
        #pragma unroll
        for (int r = 0; r < 8; ++r) {
            int m = m0 + w * 16 + half * 8 + r;
            if (m < M) C[(size_t)m * N + n] = acc[t][r] + bv;
        }
    }
}

// ---------------------------------------------------------------------------
// Elementwise / graph kernels
// ---------------------------------------------------------------------------
__global__ void zero_k(float* p, int n) {
    int i = blockIdx.x * blockDim.x + threadIdx.x;
    if (i < n) p[i] = 0.f;
}

__global__ void count_deg_k(const int* __restrict__ dst, float* __restrict__ deg, int E) {
    int e = blockIdx.x * blockDim.x + threadIdx.x;
    if (e < E) atomicAdd(&deg[dst[e]], 1.f);
}

__global__ void dinv_k(const float* __restrict__ deg, float* __restrict__ dinv, int V) {
    int v = blockIdx.x * blockDim.x + threadIdx.x;
    if (v < V) dinv[v] = rsqrtf(deg[v] + 1.f);
}

// agg[v][c] = h[v][c]*dinv[v]^2 + bias[c]     (thread per (v, 4-channel chunk))
__global__ void agg_init_k(const float* __restrict__ h, const float* __restrict__ dinv,
                           const float* __restrict__ bias, float* __restrict__ agg, int V) {
    int t = blockIdx.x * blockDim.x + threadIdx.x;
    if (t >= V * 64) return;
    int v = t >> 6, c4 = t & 63;
    float di = dinv[v];
    float s  = di * di;
    float4 hv = ((const float4*)h)[(size_t)v * 64 + c4];
    float4 bv = ((const float4*)bias)[c4];
    float4 o  = { hv.x * s + bv.x, hv.y * s + bv.y, hv.z * s + bv.z, hv.w * s + bv.w };
    ((float4*)agg)[(size_t)v * 64 + c4] = o;
}

// agg[dst] += h[src] * dinv[src]*dinv[dst]    (thread per (edge, 4-channel chunk))
__global__ void scatter_k(const int* __restrict__ src, const int* __restrict__ dst,
                          const float* __restrict__ dinv, const float* __restrict__ h,
                          float* __restrict__ agg, int E) {
    int t = blockIdx.x * blockDim.x + threadIdx.x;
    if (t >= E * 64) return;
    int e = t >> 6, c4 = t & 63;
    int s = src[e], d = dst[e];
    float coef = dinv[s] * dinv[d];
    float4 hv = ((const float4*)h)[(size_t)s * 64 + c4];
    float* ap = agg + ((size_t)d * 256 + (size_t)c4 * 4);
    atomicAdd(ap + 0, hv.x * coef);
    atomicAdd(ap + 1, hv.y * coef);
    atomicAdd(ap + 2, hv.z * coef);
    atomicAdd(ap + 3, hv.w * coef);
}

__global__ void relu_k(const float* __restrict__ agg, float* __restrict__ x, int n) {
    int i = blockIdx.x * blockDim.x + threadIdx.x;
    if (i < n) x[i] = fmaxf(agg[i], 0.f);
}

// x = (x + relu(agg)) * 0.5   (residual merge at end of each block)
__global__ void relu_res_k(const float* __restrict__ agg, float* __restrict__ x, int n) {
    int i = blockIdx.x * blockDim.x + threadIdx.x;
    if (i < n) x[i] = (x[i] + fmaxf(agg[i], 0.f)) * 0.5f;
}

// hs[v][n] = sum_c x[v][c] * W[c][n]   (N = 3, tiny)
__global__ void gemv3_k(const float* __restrict__ x, const float* __restrict__ W,
                        float* __restrict__ hs, int V) {
    int t = blockIdx.x * blockDim.x + threadIdx.x;
    if (t >= V * 3) return;
    int v = t / 3, n = t - 3 * v;
    const float* xr = x + (size_t)v * 256;
    float acc = 0.f;
    #pragma unroll 8
    for (int c = 0; c < 256; ++c) acc += xr[c] * W[c * 3 + n];
    hs[t] = acc;
}

__global__ void agg3_init_k(const float* __restrict__ hs, const float* __restrict__ dinv,
                            const float* __restrict__ b, float* __restrict__ agg3, int V) {
    int t = blockIdx.x * blockDim.x + threadIdx.x;
    if (t >= V * 3) return;
    int v = t / 3, n = t - 3 * v;
    float di = dinv[v];
    agg3[t] = hs[t] * di * di + b[n];
}

__global__ void scatter3_k(const int* __restrict__ src, const int* __restrict__ dst,
                           const float* __restrict__ dinv, const float* __restrict__ hs,
                           float* __restrict__ agg3, int E) {
    int e = blockIdx.x * blockDim.x + threadIdx.x;
    if (e >= E) return;
    int s = src[e], d = dst[e];
    float coef = dinv[s] * dinv[d];
    atomicAdd(&agg3[(size_t)d * 3 + 0], hs[(size_t)s * 3 + 0] * coef);
    atomicAdd(&agg3[(size_t)d * 3 + 1], hs[(size_t)s * 3 + 1] * coef);
    atomicAdd(&agg3[(size_t)d * 3 + 2], hs[(size_t)s * 3 + 2] * coef);
}

__global__ void copy_k(const float* __restrict__ a, float* __restrict__ o, int n) {
    int i = blockIdx.x * blockDim.x + threadIdx.x;
    if (i < n) o[i] = a[i];
}

// ---------------------------------------------------------------------------
// Host orchestration
// ---------------------------------------------------------------------------
static inline int cdiv(long long a, int b) { return (int)((a + b - 1) / b); }

extern "C" void kernel_launch(void* const* d_in, const int* in_sizes, int n_in,
                              void* d_out, int out_size, void* d_ws, size_t ws_size,
                              hipStream_t stream) {
    (void)in_sizes; (void)n_in; (void)out_size; (void)ws_size;

    constexpr int B = 4, V = 20000, E = 120000;
    constexpr int CIN = 1283, H = 256, CO = 3, NB = 6;

    const float* input = (const float*)d_in[0];
    const int*   ei    = (const int*)  d_in[1];
    const float* Wlin  = (const float*)d_in[2];
    const float* blin  = (const float*)d_in[3];
    const float* Wc1   = (const float*)d_in[4];
    const float* bc1   = (const float*)d_in[5];
    const float* Wb1   = (const float*)d_in[6];
    const float* bb1   = (const float*)d_in[7];
    const float* Wb2   = (const float*)d_in[8];
    const float* bb2   = (const float*)d_in[9];
    const float* Wc3   = (const float*)d_in[10];
    const float* bc3   = (const float*)d_in[11];

    // swizzled weight dword counts: nKb * (N/16) * 32 * 8
    const int nKbLin = cdiv(CIN, 32);                    // 41
    const int swLin  = nKbLin * (H / 16) * 32 * 8;       // 167,936 dwords
    const int swH    = (H / 32) * (H / 16) * 32 * 8;     // 32,768 dwords

    // workspace carve-up (256B aligned)
    size_t off = 0;
    auto alloc = [&](size_t bytes) -> char* {
        char* p = (char*)d_ws + off;
        off += (bytes + 255) & ~(size_t)255;
        return p;
    };
    unsigned int* wlin_sw = (unsigned int*)alloc((size_t)swLin * 4);
    unsigned int* wc1_sw  = (unsigned int*)alloc((size_t)swH * 4);
    unsigned int* wb1_sw  = (unsigned int*)alloc((size_t)NB * swH * 4);
    unsigned int* wb2_sw  = (unsigned int*)alloc((size_t)NB * swH * 4);
    float* hbuf = (float*)alloc((size_t)V * H * 4);
    float* agg  = (float*)alloc((size_t)V * H * 4);
    float* ybuf = (float*)alloc((size_t)V * H * 4);
    float* deg  = (float*)alloc((size_t)V * 4);
    float* dinv = (float*)alloc((size_t)V * 4);
    float* hs   = (float*)alloc((size_t)V * CO * 4);
    float* agg3 = (float*)alloc((size_t)V * CO * 4);

    float* xs   = (float*)d_out;                 // [B][V][H]
    float* outs = xs + (size_t)B * V * H;        // [B][V][CO]

    const int T = 256;

    // weight swizzles (recomputed every call; deterministic)
    swz_w_k<<<cdiv(swLin, T), T, 0, stream>>>(Wlin, wlin_sw, CIN, H, swLin);
    swz_w_k<<<cdiv(swH, T), T, 0, stream>>>(Wc1, wc1_sw, H, H, swH);
    for (int i = 0; i < NB; ++i) {
        swz_w_k<<<cdiv(swH, T), T, 0, stream>>>(Wb1 + (size_t)i * H * H,
                                                wb1_sw + (size_t)i * swH, H, H, swH);
        swz_w_k<<<cdiv(swH, T), T, 0, stream>>>(Wb2 + (size_t)i * H * H,
                                                wb2_sw + (size_t)i * swH, H, H, swH);
    }

    const dim3 ggrid(cdiv(V, 128), H / 64);   // 157 x 4 workgroups per GEMM

    for (int b = 0; b < B; ++b) {
        const int*   srcp = ei + (size_t)b * 2 * E;
        const int*   dstp = srcp + E;
        const float* inp  = input + (size_t)b * V * CIN;
        float*       x    = xs + (size_t)b * V * H;
        float*       outb = outs + (size_t)b * V * CO;

        // degrees (dst-only, shared across all layers of this graph)
        zero_k<<<cdiv(V, T), T, 0, stream>>>(deg, V);
        count_deg_k<<<cdiv(E, T), T, 0, stream>>>(dstp, deg, E);
        dinv_k<<<cdiv(V, T), T, 0, stream>>>(deg, dinv, V);

        // h_lin = input @ Wlin + blin   -> ybuf    (K=1283: generic path)
        gemm_v2<false><<<ggrid, T, 0, stream>>>(inp, wlin_sw, blin, ybuf, V, CIN, H);

        // x = relu(gcn(h_lin, Wc1, bc1))
        gemm_v2<true><<<ggrid, T, 0, stream>>>(ybuf, wc1_sw, nullptr, hbuf, V, H, H);
        agg_init_k<<<cdiv((long long)V * 64, T), T, 0, stream>>>(hbuf, dinv, bc1, agg, V);
        scatter_k<<<cdiv((long long)E * 64, T), T, 0, stream>>>(srcp, dstp, dinv, hbuf, agg, E);
        relu_k<<<cdiv((long long)V * H, T), T, 0, stream>>>(agg, x, V * H);

        // residual blocks
        for (int i = 0; i < NB; ++i) {
            const unsigned int* W1 = wb1_sw + (size_t)i * swH;
            const unsigned int* W2 = wb2_sw + (size_t)i * swH;
            const float* b1 = bb1 + (size_t)i * H;
            const float* b2 = bb2 + (size_t)i * H;

            // y = relu(gcn(x, W1, b1))
            gemm_v2<true><<<ggrid, T, 0, stream>>>(x, W1, nullptr, hbuf, V, H, H);
            agg_init_k<<<cdiv((long long)V * 64, T), T, 0, stream>>>(hbuf, dinv, b1, agg, V);
            scatter_k<<<cdiv((long long)E * 64, T), T, 0, stream>>>(srcp, dstp, dinv, hbuf, agg, E);
            relu_k<<<cdiv((long long)V * H, T), T, 0, stream>>>(agg, ybuf, V * H);

            // x = (x + relu(gcn(y, W2, b2))) * 0.5
            gemm_v2<true><<<ggrid, T, 0, stream>>>(ybuf, W2, nullptr, hbuf, V, H, H);
            agg_init_k<<<cdiv((long long)V * 64, T), T, 0, stream>>>(hbuf, dinv, b2, agg, V);
            scatter_k<<<cdiv((long long)E * 64, T), T, 0, stream>>>(srcp, dstp, dinv, hbuf, agg, E);
            relu_res_k<<<cdiv((long long)V * H, T), T, 0, stream>>>(agg, x, V * H);
        }

        // out = gcn(x, Wc3, bc3)  (no relu), N=3
        gemv3_k<<<cdiv((long long)V * CO, T), T, 0, stream>>>(x, Wc3, hs, V);
        agg3_init_k<<<cdiv((long long)V * CO, T), T, 0, stream>>>(hs, dinv, bc3, agg3, V);
        scatter3_k<<<cdiv(E, T), T, 0, stream>>>(srcp, dstp, dinv, hs, agg3, E);
        copy_k<<<cdiv((long long)V * CO, T), T, 0, stream>>>(agg3, outb, V * CO);
    }
}
